// DeformableConv1D_50225347559659
// MI455X (gfx1250) — matrix operand order repaired and checked by reading, not compiled, not run on hardware
//
#include <hip/hip_runtime.h>

// CDNA5 / gfx1250: wave32, WMMA 16x16x32 f16 -> f32 accumulate.
typedef __attribute__((ext_vector_type(16))) _Float16 v16h;
typedef __attribute__((ext_vector_type(4)))  _Float16 v4h;
typedef __attribute__((ext_vector_type(8)))  float    v8f;
typedef __attribute__((ext_vector_type(4)))  float    v4f;

#define N_TOTAL 4096   // 4 * 1024 * 1
#define SEQ     1024   // per-batch sequence length (conv 'same' boundary)
#define NCHUNK  (N_TOTAL / 32)   // 128 = 4 ballot groups of 32

// One wave per 16-row output tile. y = Gtot @ x with
// Gtot[i,j] = sum_k W[k] * max(0, 1 - |(j - i - off[j]) - p_k|), p = {-1,0,1}
// off[j] = relu(conv3(x)[j] + b) - x[j];  s[j] = j - off[j]  (LDS-resident)
//
// Band-skip via ballot: chunk c (32 j's) contributes to rows [i0,i0+16) only
// if some s[j] in (i0-2, i0+17) (hat support). Lane l tests chunk g*32+l;
// ballot_w32 packs a wave-uniform mask; the K-loop walks set bits with ctz,
// so dead chunks cost zero iterations and EXEC is all-1s at every WMMA.
__global__ __launch_bounds__(32) void deform_conv1d_wmma(
    const float* __restrict__ x,
    const float* __restrict__ conv_w,
    const float* __restrict__ conv_b,
    const float* __restrict__ W,
    float* __restrict__ y)
{
    __shared__ float    s_loc[N_TOTAL];  // s[j] = j - off[j]
    __shared__ _Float16 x_h[N_TOTAL];    // x in f16 (B operand)
    __shared__ float    s_min[NCHUNK];   // per-32-chunk min of s
    __shared__ float    s_max[NCHUNK];   // per-32-chunk max of s

    const int lane = threadIdx.x;        // 0..31
    const int tile = blockIdx.x;         // output rows [tile*16, tile*16+16)

    const float w0 = conv_w[0], w1 = conv_w[1], w2 = conv_w[2];
    const float cb = conv_b[0];
    const float W0 = W[0], W1 = W[1], W2 = W[2];

    // ---- Stage: 4 consecutive j per lane, b128 loads/stores. A 4-run never
    // straddles a SEQ boundary (1024 % 4 == 0), so only the run's first/last
    // element can touch the zero-padded conv edge.
    for (int j0 = lane * 4; j0 < N_TOTAL; j0 += 128) {
        const v4f xv = *(const v4f*)(x + j0);
        const int t0 = j0 & (SEQ - 1);
        float xs[6];
        xs[0] = (t0 > 0)         ? x[j0 - 1] : 0.0f;  // left neighbor
        xs[1] = xv.x; xs[2] = xv.y; xs[3] = xv.z; xs[4] = xv.w;
        xs[5] = (t0 < SEQ - 4)   ? x[j0 + 4] : 0.0f;  // right neighbor of j0+3
        v4f sv;
        v4h hv;
        #pragma unroll
        for (int e = 0; e < 4; ++e) {
            float conv = fmaf(w0, xs[e], fmaf(w1, xs[e + 1], fmaf(w2, xs[e + 2], cb)));
            conv = fmaxf(conv, 0.0f);                      // relu -> offset
            sv[e] = (float)(j0 + e) - conv + xs[e + 1];    // j - off[j]
            hv[e] = (_Float16)xs[e + 1];
        }
        *(v4f*)(s_loc + j0) = sv;
        *(v4h*)(x_h + j0)   = hv;
    }
    __syncthreads();

    // ---- Per-chunk min/max of s (4 chunks per lane, float4 LDS reads).
    for (int cidx = lane; cidx < NCHUNK; cidx += 32) {
        float mn = 3.0e38f, mx = -3.0e38f;
        const v4f* p = (const v4f*)(s_loc + cidx * 32);
        #pragma unroll
        for (int q = 0; q < 8; ++q) {
            const v4f v = p[q];
            mn = fminf(fminf(fminf(mn, v.x), v.y), fminf(v.z, v.w));
            mx = fmaxf(fmaxf(fmaxf(mx, v.x), v.y), fmaxf(v.z, v.w));
        }
        s_min[cidx] = mn;
        s_max[cidx] = mx;
    }
    __syncthreads();

    const int   m    = lane & 15;        // output row within tile / B column
    const int   half = lane >> 4;        // 0 or 1 (K-half selector)
    const float t16  = (float)(tile * 16);
    const float i_f  = t16 + (float)m;

    // ---- Ballot-compressed active-chunk masks (wave-uniform).
    // Boundary is exact: at distance 2 the hat weight is exactly 0.
    unsigned int cmask[4];
    #pragma unroll
    for (int g = 0; g < 4; ++g) {
        const int  cidx   = g * 32 + lane;
        const bool active = !(s_max[cidx] <= t16 - 2.0f ||
                              s_min[cidx] >= t16 + 17.0f);
        cmask[g] = __builtin_amdgcn_ballot_w32(active);
    }

    // B-matrix (32x16 f16): only column n==0 is nonzero -> lanes 0 and 16.
    v16h b;
    #pragma unroll
    for (int q = 0; q < 16; ++q) b[q] = (_Float16)0.0f;

    v8f c = {};  // f32 accumulator, D[m,n]
    #pragma unroll
    for (int g = 0; g < 4; ++g) {
        unsigned int mask = cmask[g];
        while (mask) {
            const int bit = __builtin_ctz(mask);
            mask &= mask - 1;
            const int jb = (g * 32 + bit) * 32;   // uniform scalar chunk base

            // A-matrix (16x32 f16), ISA layout:
            //   lanes 0-15 : V0..3 -> K=0..7,  V4..7 -> K=16..23
            //   lanes 16-31: V0..3 -> K=8..15, V4..7 -> K=24..31
            v16h a;
            #pragma unroll
            for (int v = 0; v < 8; ++v) {
                #pragma unroll
                for (int e = 0; e < 2; ++e) {
                    const int   kk = ((v & 4) ? 16 : 0) + ((v & 3) * 2 + e) + half * 8;
                    const float D  = s_loc[jb + kk] - i_f;       // j - i - off[j]
                    float g2 = W1 * fmaxf(0.0f, 1.0f - fabsf(D));
                    g2 = fmaf(W0, fmaxf(0.0f, 1.0f - fabsf(D + 1.0f)), g2);
                    g2 = fmaf(W2, fmaxf(0.0f, 1.0f - fabsf(D - 1.0f)), g2);
                    a[v * 2 + e] = (_Float16)g2;
                }
            }

            if (m == 0) {
                #pragma unroll
                for (int q = 0; q < 16; ++q) b[q] = x_h[jb + ((q & 8) ? 16 : 0) + half * 8 + (q & 7)];
            }

            // D = A x B + C ; emits v_wmma_f32_16x16x32_f16
            c = __builtin_amdgcn_wmma_f32_16x16x32_f16(
                    /*neg_a=*/false, a, /*neg_b=*/false, b,
                    /*c_mod=*/(short)0, c, /*reuse_a=*/false, /*reuse_b=*/false);
        }
    }

    // D layout: VGPR r -> (M=r, n=lane) for lanes 0-15, (M=r+8, n=lane-16).
    // y is column n==0 -> lane 0 writes rows 0..7, lane 16 writes rows 8..15.
    if (m == 0) {
        const int base = tile * 16 + half * 8;
        #pragma unroll
        for (int r = 0; r < 8; ++r) y[base + r] = c[r];
    }
}

extern "C" void kernel_launch(void* const* d_in, const int* in_sizes, int n_in,
                              void* d_out, int out_size, void* d_ws, size_t ws_size,
                              hipStream_t stream) {
    (void)in_sizes; (void)n_in; (void)d_ws; (void)ws_size; (void)out_size;
    const float* x  = (const float*)d_in[0];  // (4,1024,1)
    const float* cw = (const float*)d_in[1];  // (3,1,1)
    const float* cb = (const float*)d_in[2];  // (1,)
    const float* W  = (const float*)d_in[3];  // (3,1)
    float* y = (float*)d_out;                 // (4,1024,1) flat = 4096
    deform_conv1d_wmma<<<N_TOTAL / 16, 32, 0, stream>>>(x, cw, cb, W, y);
}
